// RLayer_80942953661002
// MI455X (gfx1250) — compile-verified, matrix operand order checked by reading, not run on hardware
//
#include <hip/hip_runtime.h>
#include <hip/hip_bf16.h>

// ---------------- problem constants ----------------
#define T_STEPS 256
#define BATCH   256
#define INP_DIM 64
#define HID     1024
#define KCAT    (HID + INP_DIM)   // 1088 concatenated K (recurrent + input proj)
#define ALPHA_F 0.05f
#define ONE_MINUS_ALPHA_F 0.95f

typedef __attribute__((ext_vector_type(16))) __bf16 v16bf;
typedef __attribute__((ext_vector_type(8)))  __bf16 v8bf;
typedef __attribute__((ext_vector_type(4)))  __bf16 v4bf;
typedef __attribute__((ext_vector_type(8)))  float  v8f;

// TDM descriptor vector types (6-arg builtin on this toolchain)
typedef __attribute__((ext_vector_type(4))) unsigned int tdm_v4u;
typedef __attribute__((ext_vector_type(8))) int          tdm_v8i;
typedef __attribute__((ext_vector_type(4))) int          tdm_v4i;

// LDS row stride (elements): 64 data + 8 pad -> 144B rows, conflict-free for
// 16-lane b128 fragment reads. Matches TDM pad: 32 DWORDs data + 4 DWORDs pad.
#define LDSTR 72

// ---------------------------------------------------------------------------
// Kernel 0: pack Wcat = [W_rec | W_in] as bf16 [HID][KCAT], fuse biases,
// zero the x-state and the r0 (= relu(0) = 0) buffer.
// ---------------------------------------------------------------------------
__global__ __launch_bounds__(256) void rnn_pack_kernel(
    const float* __restrict__ W_in,      // [HID, INP_DIM]
    const float* __restrict__ bias_in,   // [HID]
    const float* __restrict__ W_rec,     // [HID, HID]
    const float* __restrict__ bias_rec,  // [HID]
    __bf16* __restrict__ Wcat,           // [HID, KCAT] bf16
    float*  __restrict__ biasSum,        // [HID]
    float*  __restrict__ xState,         // [BATCH, HID]
    float*  __restrict__ r0)             // [BATCH, HID]
{
    const int stride = gridDim.x * blockDim.x;
    const int total  = HID * KCAT;
    for (int idx = blockIdx.x * blockDim.x + threadIdx.x; idx < total; idx += stride) {
        const int h = idx / KCAT;
        const int k = idx - h * KCAT;
        const float v = (k < HID) ? W_rec[(size_t)h * HID + k]
                                  : W_in[(size_t)h * INP_DIM + (k - HID)];
        Wcat[idx] = (__bf16)v;
        if (idx < HID)          biasSum[idx] = bias_in[idx] + bias_rec[idx];
        if (idx < BATCH * HID)  { xState[idx] = 0.0f; r0[idx] = 0.0f; }
    }
}

// ---------------------------------------------------------------------------
// Kernel 1: one recurrence step (double-buffered, TDM weight streaming).
//   acc[b,h] = sum_k [r_{t-1} | u_t](b,k) * Wcat(h,k)      (bf16 WMMA, f32 acc)
//   x1 = 0.95*x + 0.05*(acc + biasSum)   ;   out = relu(x1)
// ---------------------------------------------------------------------------
__global__ __launch_bounds__(128) void rnn_step_kernel(
    const float*  __restrict__ rPrev,    // [BATCH, HID]
    const float*  __restrict__ inpT,     // [BATCH, INP_DIM]
    const __bf16* __restrict__ Wcat,     // [HID, KCAT]
    const float*  __restrict__ biasSum,  // [HID]
    float*        __restrict__ xState,   // [BATCH, HID]
    float*        __restrict__ outT)     // [BATCH, HID]
{
    __shared__ __bf16 lA[2][64 * LDSTR];   // activation tiles (rows = batch)
    __shared__ __bf16 lB[2][64 * LDSTR];   // weight tiles     (rows = hidden col)

    const int tid      = threadIdx.x;
    const int lane     = tid & 31;
    const int wave     = tid >> 5;       // 0..3  (wave32)
    const int laneHalf = lane >> 4;      // 0 | 1
    const int lr       = lane & 15;      // row/col within 16
    const int colBase  = blockIdx.x * 64;    // hidden tile
    const int rowBase  = blockIdx.y * 64;    // batch tile
    const int mBase    = wave * 16;          // wave's 16-row strip

    v8f acc[4] = {v8f{}, v8f{}, v8f{}, v8f{}};
    float4 regs[8];

    // ---- stage A chunk kc into registers (8 x float4 per thread) ----
    auto loadA = [&](int kc) {
        const float* src  = (kc < 16) ? rPrev : inpT;
        const int strideS = (kc < 16) ? HID : INP_DIM;
        const int kOff    = (kc < 16) ? kc * 64 : 0;
        #pragma unroll
        for (int it = 0; it < 8; ++it) {
            const int i   = tid + it * 128;      // 0..1023, compile-time bound
            const int row = i >> 4;
            const int kv  = (i & 15) * 4;
            regs[it] = *(const float4*)&src[(size_t)(rowBase + row) * strideS + kOff + kv];
        }
    };
    // ---- convert + store staged A registers into LDS buffer ----
    auto storeA = [&](int buf) {
        #pragma unroll
        for (int it = 0; it < 8; ++it) {
            const int i   = tid + it * 128;
            const int row = i >> 4;
            const int kv  = (i & 15) * 4;
            v4bf o = { (__bf16)regs[it].x, (__bf16)regs[it].y,
                       (__bf16)regs[it].z, (__bf16)regs[it].w };
            *(v4bf*)&lA[buf][row * LDSTR + kv] = o;
        }
    };
    // ---- TDM: async load 64x64 bf16 weight tile into LDS (wave 0 issues) ----
    auto issueB = [&](int kc, int buf) {
        if (wave != 0) return;
        const unsigned lds = (unsigned)(size_t)&lB[buf][0];
        const unsigned long long ga =
            (unsigned long long)(size_t)&Wcat[(size_t)colBase * KCAT + kc * 64];
        // D# group 0: count=1 | lds_addr | global_addr | type=2
        tdm_v4u g0 = { 1u,
                       lds,
                       (unsigned)(ga & 0xFFFFFFFFull),
                       (unsigned)((ga >> 32) & 0x1FFFFFFull) | (2u << 30) };
        // D# group 1: data_size=2B | pad_enable | pad_interval=32dw | pad_amount=4dw
        //             tensor_dim0=KCAT, tensor_dim1=HID, tile 64x64, stride0=KCAT
        tdm_v8i g1 = {
            (int)((1u << 16) | (1u << 20) | (4u << 22) | (3u << 25)),
            (int)(((unsigned)KCAT & 0xFFFFu) << 16),
            (int)((((unsigned)KCAT >> 16) & 0xFFFFu) | (((unsigned)HID & 0xFFFFu) << 16)),
            (int)((((unsigned)HID >> 16) & 0xFFFFu) | (64u << 16)),
            (int)64,
            (int)KCAT,
            0, 0 };
        tdm_v4i g2 = { 0, 0, 0, 0 };
        tdm_v4i g3 = { 0, 0, 0, 0 };
        tdm_v8i g4 = { 0, 0, 0, 0, 0, 0, 0, 0 };
        __builtin_amdgcn_tensor_load_to_lds(g0, g1, g2, g3, g4, 0);
    };
    // ---- 8 WMMAs over one staged 64-wide K chunk ----
    auto compute = [&](int buf) {
        const __bf16* A = &lA[buf][0];
        const __bf16* B = &lB[buf][0];
        #pragma unroll
        for (int ks = 0; ks < 64; ks += 32) {
            const int ko = ks + laneHalf * 8;  // K layout {0..7,16..23}/{8..15,24..31}
            union { v16bf v; v8bf h[2]; } afr;
            const __bf16* ar = &A[(mBase + lr) * LDSTR + ko];
            afr.h[0] = *(const v8bf*)(ar);
            afr.h[1] = *(const v8bf*)(ar + 16);
            #pragma unroll
            for (int n = 0; n < 4; ++n) {
                union { v16bf v; v8bf h[2]; } bfr;
                const __bf16* br = &B[(n * 16 + lr) * LDSTR + ko];
                bfr.h[0] = *(const v8bf*)(br);
                bfr.h[1] = *(const v8bf*)(br + 16);
                acc[n] = __builtin_amdgcn_wmma_f32_16x16x32_bf16(
                    false, afr.v, false, bfr.v, (short)0, acc[n], false, false);
            }
        }
    };

    // ---- prologue: stage chunk 0 into buffer 0 ----
    loadA(0);
    issueB(0, 0);
    storeA(0);
    if (wave == 0) __builtin_amdgcn_s_wait_tensorcnt(0);
    __syncthreads();

    // ---- pipelined main loop: compute kc while streaming kc+1 ----
    for (int kc = 0; kc < KCAT / 64; ++kc) {       // 17 chunks
        const int cur = kc & 1;
        const int nxt = cur ^ 1;
        if (kc < 16) {
            issueB(kc + 1, nxt);     // TDM streams next weight slab
            loadA(kc + 1);           // next activation slab -> registers
        }
        compute(cur);                // 8 x v_wmma on current buffers
        if (kc < 16) {
            storeA(nxt);             // cvt f32->bf16, park in next buffer
            if (wave == 0) __builtin_amdgcn_s_wait_tensorcnt(0);
            __syncthreads();
        }
    }

    // ---- epilogue: leaky integrate + ReLU ----
    // D layout: vgpr r -> M = r + 8*laneHalf ; N = lane%16
    const int bRow = rowBase + mBase + laneHalf * 8;
    #pragma unroll
    for (int n = 0; n < 4; ++n) {
        const int gh = colBase + n * 16 + lr;
        const float bias = biasSum[gh];
        #pragma unroll
        for (int r = 0; r < 8; ++r) {
            const size_t off = (size_t)(bRow + r) * HID + gh;
            const float x1 = ONE_MINUS_ALPHA_F * xState[off]
                           + ALPHA_F * (acc[n][r] + bias);
            xState[off] = x1;
            outT[off]   = fmaxf(x1, 0.0f);
        }
    }
}

// ---------------------------------------------------------------------------
extern "C" void kernel_launch(void* const* d_in, const int* in_sizes, int n_in,
                              void* d_out, int out_size, void* d_ws, size_t ws_size,
                              hipStream_t stream) {
    const float* inp      = (const float*)d_in[0];  // [T,B,INP]
    const float* W_in     = (const float*)d_in[1];  // [H,INP]
    const float* bias_in  = (const float*)d_in[2];  // [H]
    const float* W_rec    = (const float*)d_in[3];  // [H,H]
    const float* bias_rec = (const float*)d_in[4];  // [H]
    float* out = (float*)d_out;                     // [T,B,H]

    // workspace layout (16B-aligned)
    char* ws = (char*)d_ws;
    __bf16* Wcat    = (__bf16*)(ws);                            // 2,228,224 B
    float*  biasSum = (float*)(ws + 2228224);                   //     4,096 B
    float*  xState  = (float*)(ws + 2228224 + 4096);            // 1,048,576 B
    float*  r0      = (float*)(ws + 2228224 + 4096 + 1048576);  // 1,048,576 B

    rnn_pack_kernel<<<1024, 256, 0, stream>>>(W_in, bias_in, W_rec, bias_rec,
                                              Wcat, biasSum, xState, r0);

    const dim3 grid(HID / 64, BATCH / 64);   // 16 x 4 tiles
    for (int t = 0; t < T_STEPS; ++t) {
        const float* rP = (t == 0) ? r0 : out + (size_t)(t - 1) * BATCH * HID;
        rnn_step_kernel<<<grid, 128, 0, stream>>>(
            rP,
            inp + (size_t)t * BATCH * INP_DIM,
            Wcat, biasSum, xState,
            out + (size_t)t * BATCH * HID);
    }
}
